// CollaborativeExpertsWrapper_32770600468830
// MI455X (gfx1250) — compile-verified, matrix operand order checked by reading, not run on hardware
//
#include <hip/hip_runtime.h>
#include <hip/hip_bf16.h>
#include <math.h>

// ---------------------------------------------------------------------------
// Types for CDNA5 WMMA (wave32): v_wmma_f32_16x16x32_bf16
// A: 16x32 bf16 -> 16 bf16 per lane (8 VGPRs). B: 32x16 bf16 symmetric.
// C/D: 16x16 f32 -> 8 f32 per lane.
// ---------------------------------------------------------------------------
typedef __bf16 bf16_t;
typedef __attribute__((ext_vector_type(16))) __bf16 v16bf;
typedef __attribute__((ext_vector_type(8)))  float  v8f;

union FragBF { v16bf v; uint4 q[2]; };

// Load one 16-lane-striped WMMA operand fragment from a row-major bf16 source.
// Caller passes p = rowBase + 8*(lane>>4); chunks live at +0 and +16 elements.
__device__ __forceinline__ v16bf load_frag(const bf16_t* p) {
  FragBF f;
  f.q[0] = *reinterpret_cast<const uint4*>(p);
  f.q[1] = *reinterpret_cast<const uint4*>(p + 16);
  return f.v;
}

// ---------------------------------------------------------------------------
// Elementwise f32 -> bf16
// ---------------------------------------------------------------------------
__global__ void k_cvt_bf16(const float* __restrict__ src, bf16_t* __restrict__ dst, int n) {
  int i = blockIdx.x * blockDim.x + threadIdx.x;
  if (i < n) dst[i] = (bf16_t)src[i];
}

// Transpose-convert weight: src f32 (K x N) row-major -> dst bf16 (N x K).
// Lets the GEMM's B operand (lane = output column) load contiguously.
__global__ void k_transpose_cvt(const float* __restrict__ src, bf16_t* __restrict__ dst,
                                int K, int N) {
  int i = blockIdx.x * blockDim.x + threadIdx.x; // dst linear index = n*K + k
  if (i >= K * N) return;
  int n = i / K;
  int k = i - n * K;
  dst[i] = (bf16_t)src[(size_t)k * N + n];
}

// ---------------------------------------------------------------------------
// Generic WMMA GEMM:  C(MxN) = A(MxK) @ Bt(NxK)^T, +bias, *scale
// One wave per 16x64 strip (4 n-tiles): one A fragment feeds 4 back-to-back
// WMMAs per k-step (4x less A traffic, WMMA-burst friendly). 8 waves/block.
// Requires N % 64 == 0 (holds for all call sites: 512, 1024, 128).
// MODE 0: f32 row-major out.  MODE 1: bf16 row-major out (ldc/colOff for
// writing into a wider feat row).  MODE 2: bf16 head-transposed V output
// Vt[((b*4+h)*128 + d)*64 + t] for the attention P@V operand.
// ---------------------------------------------------------------------------
template <int MODE>
__global__ void k_gemm_wmma(const bf16_t* __restrict__ A, int lda,
                            const bf16_t* __restrict__ Bt, int ldb,
                            const float* __restrict__ bias,
                            void* __restrict__ Cout, int ldc, int colOff,
                            int M, int N, int K, float scale)
{
  int wave = blockIdx.x * (blockDim.x >> 5) + (threadIdx.x >> 5);
  int lane = threadIdx.x & 31;
  int nStripsN = N >> 6;                      // 4 n-tiles per strip
  int nStrips  = (M >> 4) * nStripsN;
  if (wave >= nStrips) return;                // wave-uniform: EXEC stays all-ones
  int tm  = wave / nStripsN;
  int tn0 = (wave - tm * nStripsN) * 4;
  int hl = lane >> 4, ln = lane & 15;

  const bf16_t* arow = A + (size_t)(tm * 16 + ln) * lda + hl * 8;
  const bf16_t* brow[4];
#pragma unroll
  for (int j = 0; j < 4; ++j)
    brow[j] = Bt + (size_t)((tn0 + j) * 16 + ln) * ldb + hl * 8;

  v8f acc0 = {}, acc1 = {}, acc2 = {}, acc3 = {};
  for (int k0 = 0; k0 < K; k0 += 32) {
    v16bf af = load_frag(arow + k0);
    v16bf b0 = load_frag(brow[0] + k0);
    v16bf b1 = load_frag(brow[1] + k0);
    v16bf b2 = load_frag(brow[2] + k0);
    v16bf b3 = load_frag(brow[3] + k0);
    acc0 = __builtin_amdgcn_wmma_f32_16x16x32_bf16(false, af, false, b0, (short)0, acc0, false, false);
    acc1 = __builtin_amdgcn_wmma_f32_16x16x32_bf16(false, af, false, b1, (short)0, acc1, false, false);
    acc2 = __builtin_amdgcn_wmma_f32_16x16x32_bf16(false, af, false, b2, (short)0, acc2, false, false);
    acc3 = __builtin_amdgcn_wmma_f32_16x16x32_bf16(false, af, false, b3, (short)0, acc3, false, false);
  }

  v8f accs[4] = { acc0, acc1, acc2, acc3 };
#pragma unroll
  for (int j = 0; j < 4; ++j) {
    int ncol = (tn0 + j) * 16 + ln;
    float bv = bias ? bias[ncol] : 0.0f;
#pragma unroll
    for (int r = 0; r < 8; ++r) {
      int mrow = tm * 16 + r + 8 * hl;
      float v = (accs[j][r] + bv) * scale;
      if (MODE == 0) {
        ((float*)Cout)[(size_t)mrow * ldc + colOff + ncol] = v;
      } else if (MODE == 1) {
        ((bf16_t*)Cout)[(size_t)mrow * ldc + colOff + ncol] = (bf16_t)v;
      } else {
        int bb = mrow >> 6, t = mrow & 63;   // row = b*64 + t
        int hh = ncol >> 7, d = ncol & 127;  // col = h*128 + d
        ((bf16_t*)Cout)[((size_t)(bb * 4 + hh) * 128 + d) * 64 + t] = (bf16_t)v;
      }
    }
  }
}

// ---------------------------------------------------------------------------
// Attention for one (batch, head): scores = (Q/sqrt(hd)) @ K^T (64x64),
// softmax in LDS, then P @ V with V pre-transposed per head. 4 waves, each
// owns a 16-row strip. Q already carries the 1/sqrt(128) scale.
// ---------------------------------------------------------------------------
__global__ void k_attention(const bf16_t* __restrict__ Q, const bf16_t* __restrict__ Kb,
                            const bf16_t* __restrict__ Vt, bf16_t* __restrict__ Out)
{
  __shared__ float  sc[64][68];
  __shared__ bf16_t pr[64][64];
  int bh = blockIdx.x, b = bh >> 2, h = bh & 3;
  int w = threadIdx.x >> 5, lane = threadIdx.x & 31;
  int hl = lane >> 4, ln = lane & 15;

  // --- scores: S = Q_tile @ K^T, K-dim = 128 (4 k-tiles) ---
  const bf16_t* qrow = Q + (size_t)(b * 64 + w * 16 + ln) * 512 + h * 128 + hl * 8;
  for (int nt = 0; nt < 4; ++nt) {
    const bf16_t* krow = Kb + (size_t)(b * 64 + nt * 16 + ln) * 512 + h * 128 + hl * 8;
    v8f acc = {};
#pragma unroll
    for (int kt = 0; kt < 4; ++kt) {
      v16bf af = load_frag(qrow + kt * 32);
      v16bf bf = load_frag(krow + kt * 32);
      acc = __builtin_amdgcn_wmma_f32_16x16x32_bf16(false, af, false, bf,
                                                    (short)0, acc, false, false);
    }
#pragma unroll
    for (int r = 0; r < 8; ++r) sc[w * 16 + r + 8 * hl][nt * 16 + ln] = acc[r];
  }
  __syncthreads();

  // --- softmax over each of 64 rows (threads 0..63 = waves 0,1: uniform) ---
  if (threadIdx.x < 64) {
    int row = threadIdx.x;
    float mx = -3.0e38f;
    for (int j = 0; j < 64; ++j) mx = fmaxf(mx, sc[row][j]);
    float s = 0.0f;
    for (int j = 0; j < 64; ++j) { float e = __expf(sc[row][j] - mx); sc[row][j] = e; s += e; }
    float inv = 1.0f / s;
    for (int j = 0; j < 64; ++j) pr[row][j] = (bf16_t)(sc[row][j] * inv);
  }
  __syncthreads();

  // --- out = P @ V, K-dim = 64 (2 k-tiles), N = 128 (8 n-tiles) ---
  const bf16_t* prow = &pr[w * 16 + ln][0] + hl * 8;
  for (int nt = 0; nt < 8; ++nt) {
    const bf16_t* vrow = Vt + ((size_t)(b * 4 + h) * 128 + nt * 16 + ln) * 64 + hl * 8;
    v8f acc = {};
#pragma unroll
    for (int kt = 0; kt < 2; ++kt) {
      v16bf af = load_frag(prow + kt * 32);
      v16bf bf = load_frag(vrow + kt * 32);
      acc = __builtin_amdgcn_wmma_f32_16x16x32_bf16(false, af, false, bf,
                                                    (short)0, acc, false, false);
    }
#pragma unroll
    for (int r = 0; r < 8; ++r)
      Out[(size_t)(b * 64 + w * 16 + r + 8 * hl) * 512 + h * 128 + nt * 16 + ln] =
          (bf16_t)acc[r];
  }
}

// ---------------------------------------------------------------------------
// Mean over the middle (token) axis: src f32 (Bt x T x C) -> dst bf16 (Bt x C)
// Coalesced streaming; global_prefetch keeps the pipe ahead (speculative,
// out-of-range tail prefetches are silently dropped per the ISA).
// ---------------------------------------------------------------------------
__global__ void k_mean_rows(const float* __restrict__ src, bf16_t* __restrict__ dst,
                            int T, int C, float inv) {
  int b = blockIdx.x;
  for (int c = threadIdx.x; c < C; c += blockDim.x) {
    const float* p = src + (size_t)b * T * C + c;
    float s = 0.0f;
    for (int t = 0; t < T; ++t) {
      __builtin_prefetch(p + (size_t)(t + 16) * C, 0, 0);
      s += p[(size_t)t * C];
    }
    dst[(size_t)b * C + c] = (bf16_t)(s * inv);
  }
}

// Row L2-normalize: src f32 (B x C) -> dst bf16 (B x C)
__global__ void k_rownorm(const float* __restrict__ src, bf16_t* __restrict__ dst, int C) {
  __shared__ float red[256];
  int b = blockIdx.x;
  float s = 0.0f;
  for (int c = threadIdx.x; c < C; c += blockDim.x) {
    float v = src[(size_t)b * C + c];
    s += v * v;
  }
  red[threadIdx.x] = s;
  __syncthreads();
  for (int off = blockDim.x >> 1; off > 0; off >>= 1) {
    if ((int)threadIdx.x < off) red[threadIdx.x] += red[threadIdx.x + off];
    __syncthreads();
  }
  float inv = 1.0f / fmaxf(sqrtf(red[0]), 1e-12f);
  for (int c = threadIdx.x; c < C; c += blockDim.x)
    dst[(size_t)b * C + c] = (bf16_t)(src[(size_t)b * C + c] * inv);
}

// ---------------------------------------------------------------------------
// Both ranking directions from the Gram matrix G[a][c] = oo[a].feat[c].
//   dir 0: s[i][b] = G[b][(b-i)%B]   (x1=oo, x2=feat)
//   dir 1: s[i][b] = G[(b-i)%B][b]   (x1=feat, x2=oo)
// One block, 128 threads (thread = column b). out[0]=loss, out[1]=acc.
// ---------------------------------------------------------------------------
__global__ void k_ranking(const float* __restrict__ G, const int* __restrict__ gm,
                          float* __restrict__ out) {
  __shared__ float rA[128], rB[128], rC[128], rD[128];
  __shared__ float partial[2][2];
  int b = threadIdx.x;
  bool gb = gm[b] != 0;
  for (int dir = 0; dir < 2; ++dir) {
    float pos = G[b * 128 + b];   // s[0][b] in both directions
    float lossSum = 0.0f, selSum = 0.0f, smax = -3.0e38f;
    int   cnt = 0;
    bool  anym = false;
    for (int i = 0; i < 128; ++i) {
      int   j  = (b - i) & 127;
      float sv = (dir == 0) ? G[b * 128 + j] : G[j * 128 + b];
      bool  mk = gb && (gm[j] != 0);
      if (mk) {
        anym = true;
        selSum += sv;
        smax = fmaxf(smax, sv);
        if (i > 0) { ++cnt; lossSum += fmaxf(0.0f, 1.0f + sv - pos); }
      }
    }
    float colMean  = (cnt > 0) ? lossSum / (float)cnt : 0.0f;
    float validF   = (cnt > 0) ? 1.0f : 0.0f;
    bool  colValid = (selSum != 0.0f) && anym;
    float topFirst = (pos >= smax) ? 1.0f : 0.0f;
    rA[b] = colMean * validF;
    rB[b] = validF;
    rC[b] = colValid ? topFirst : 0.0f;
    rD[b] = colValid ? 1.0f : 0.0f;
    __syncthreads();
    for (int off = 64; off > 0; off >>= 1) {
      if (b < off) { rA[b] += rA[b+off]; rB[b] += rB[b+off];
                     rC[b] += rC[b+off]; rD[b] += rD[b+off]; }
      __syncthreads();
    }
    if (b == 0) {
      partial[dir][0] = rA[0] / fmaxf(rB[0], 1.0f);
      partial[dir][1] = rC[0] / fmaxf(rD[0], 1.0f);
    }
    __syncthreads();
  }
  if (b == 0) {
    out[0] = partial[0][0] + partial[1][0];
    out[1] = 0.5f * (partial[0][1] + partial[1][1]);
  }
}

// ---------------------------------------------------------------------------
// Orchestration
// ---------------------------------------------------------------------------
extern "C" void kernel_launch(void* const* d_in, const int* in_sizes, int n_in,
                              void* d_out, int out_size, void* d_ws, size_t ws_size,
                              hipStream_t stream)
{
  (void)in_sizes; (void)n_in; (void)out_size; (void)ws_size;
  const float* o     = (const float*)d_in[0];
  const float* rgb   = (const float*)d_in[1];
  const float* audio = (const float*)d_in[2];
  const int*   gmask = (const int*)d_in[3];
  const float* W[2][5];  // [m][0..4] = Wq, Wk, Wv, Wo, W2
  const float* Bi[2][5]; // biases
  for (int m = 0; m < 2; ++m) {
    int base = 4 + m * 10;
    for (int k = 0; k < 5; ++k) {
      W[m][k]  = (const float*)d_in[base + 2 * k];
      Bi[m][k] = (const float*)d_in[base + 2 * k + 1];
    }
  }
  const float* expandW = (const float*)d_in[24];
  const float* expandB = (const float*)d_in[25];

  size_t off = 0;
  auto alloc = [&](size_t bytes) -> void* {
    off = (off + 255) & ~(size_t)255;
    void* p = (char*)d_ws + off;
    off += bytes;
    return p;
  };
  const int B = 128, T = 64, D = 512, H = 4;
  const int R = B * T; // 8192 token rows

  bf16_t* xbf    = (bf16_t*)alloc((size_t)R * 2048 * 2);   // activations (rgb/audio reuse)
  bf16_t* wqt    = (bf16_t*)alloc((size_t)512 * 2048 * 2); // transposed weights (reused)
  bf16_t* wkt    = (bf16_t*)alloc((size_t)512 * 2048 * 2);
  bf16_t* wvt    = (bf16_t*)alloc((size_t)512 * 2048 * 2);
  bf16_t* wot    = (bf16_t*)alloc((size_t)512 * 512 * 2);
  bf16_t* w2t    = (bf16_t*)alloc((size_t)512 * 512 * 2);
  bf16_t* Qbf    = (bf16_t*)alloc((size_t)R * D * 2);
  bf16_t* Kbf    = (bf16_t*)alloc((size_t)R * D * 2);
  bf16_t* Vt     = (bf16_t*)alloc((size_t)R * D * 2);      // (B*H, 128, 64) head-transposed
  bf16_t* attn   = (bf16_t*)alloc((size_t)R * D * 2);
  float*  h1     = (float*) alloc((size_t)R * D * 4);
  bf16_t* pooled = (bf16_t*)alloc((size_t)B * D * 2);
  bf16_t* featbf = (bf16_t*)alloc((size_t)B * 1024 * 2);
  bf16_t* ombf   = (bf16_t*)alloc((size_t)B * D * 2);
  bf16_t* expWt  = (bf16_t*)alloc((size_t)1024 * 512 * 2);
  float*  oof    = (float*) alloc((size_t)B * 1024 * 4);
  bf16_t* oobf   = (bf16_t*)alloc((size_t)B * 1024 * 2);
  float*  G      = (float*) alloc((size_t)B * B * 4);

  const float qscale = 0.08838834764831845f; // 1/sqrt(128)
  auto gemmBlocks = [](int M, int N) -> unsigned {
    return (unsigned)(((M >> 4) * (N >> 6) + 7) / 8); // 16x64 strips, 8 waves/block
  };

  const float* xs[2] = { rgb, audio };
  const int    ds[2] = { 2048, 128 };
  for (int m = 0; m < 2; ++m) {
    int d  = ds[m];
    int nx = R * d;
    k_cvt_bf16<<<(nx + 255) / 256, 256, 0, stream>>>(xs[m], xbf, nx);
    k_transpose_cvt<<<(d * 512 + 255) / 256, 256, 0, stream>>>(W[m][0], wqt, d, 512);
    k_transpose_cvt<<<(d * 512 + 255) / 256, 256, 0, stream>>>(W[m][1], wkt, d, 512);
    k_transpose_cvt<<<(d * 512 + 255) / 256, 256, 0, stream>>>(W[m][2], wvt, d, 512);
    k_transpose_cvt<<<(512 * 512 + 255) / 256, 256, 0, stream>>>(W[m][3], wot, 512, 512);
    k_transpose_cvt<<<(512 * 512 + 255) / 256, 256, 0, stream>>>(W[m][4], w2t, 512, 512);

    k_gemm_wmma<1><<<gemmBlocks(R, 512), 256, 0, stream>>>(
        xbf, d, wqt, d, Bi[m][0], Qbf, 512, 0, R, 512, d, qscale);
    k_gemm_wmma<1><<<gemmBlocks(R, 512), 256, 0, stream>>>(
        xbf, d, wkt, d, Bi[m][1], Kbf, 512, 0, R, 512, d, 1.0f);
    k_gemm_wmma<2><<<gemmBlocks(R, 512), 256, 0, stream>>>(
        xbf, d, wvt, d, Bi[m][2], Vt, 0, 0, R, 512, d, 1.0f);

    k_attention<<<B * H, 128, 0, stream>>>(Qbf, Kbf, Vt, attn);

    k_gemm_wmma<0><<<gemmBlocks(R, 512), 256, 0, stream>>>(
        attn, 512, wot, 512, Bi[m][3], h1, 512, 0, R, 512, 512, 1.0f);
    k_mean_rows<<<B, 256, 0, stream>>>(h1, pooled, T, 512, 1.0f / 64.0f);
    k_gemm_wmma<1><<<gemmBlocks(B, 512), 256, 0, stream>>>(
        pooled, 512, w2t, 512, Bi[m][4], featbf, 1024, m * 512, B, 512, 512, 1.0f);
  }

  // oo path: mean over 1024 tokens of o (the 268MB stream), expand, normalize
  k_mean_rows<<<B, 512, 0, stream>>>(o, ombf, 1024, 512, 1.0f / 1024.0f);
  k_transpose_cvt<<<(512 * 1024 + 255) / 256, 256, 0, stream>>>(expandW, expWt, 512, 1024);
  k_gemm_wmma<0><<<gemmBlocks(B, 1024), 256, 0, stream>>>(
      ombf, 512, expWt, 512, expandB, oof, 1024, 0, B, 1024, 512, 1.0f);
  k_rownorm<<<B, 256, 0, stream>>>(oof, oobf, 1024);

  // Gram matrix G = oo @ feat^T (feat rows are exactly the B-operand layout)
  k_gemm_wmma<0><<<gemmBlocks(B, B), 256, 0, stream>>>(
      oobf, 1024, featbf, 1024, (const float*)nullptr, G, 128, 0, B, B, 1024, 1.0f);

  k_ranking<<<1, 128, 0, stream>>>(G, gmask, (float*)d_out);
}
